// EncoderDecoder_56513179681102
// MI455X (gfx1250) — matrix operand order repaired and checked by reading, not compiled, not run on hardware
//
#include <hip/hip_runtime.h>
#include <hip/hip_bf16.h>
#include <cstdint>

typedef __bf16 bf16;
typedef __attribute__((ext_vector_type(8)))  __bf16 bf16x8;   // 16 B
typedef __attribute__((ext_vector_type(16))) __bf16 v16bf;    // 32 B WMMA A/B operand
typedef __attribute__((ext_vector_type(8)))  float  v8f;      // WMMA C/D operand

#define BM 128
#define BN 64
#define BK 64
#define LDSA (BK + 8)   // +8 bf16 (16B) pad -> row stride 144B, conflict-free b128 reads

// Direct global->LDS async copy path (ASYNCcnt-tracked, no VGPR staging).
// Gated on builtin availability; falls back to reg-staged pipeline otherwise.
#if defined(__HIP_DEVICE_COMPILE__) && __has_builtin(__builtin_amdgcn_global_load_async_to_lds_b128)
#define USE_ASYNC_COPY 1
#else
#define USE_ASYNC_COPY 0
#endif

#if USE_ASYNC_COPY
typedef int v4i __attribute__((vector_size(16)));   // b128 payload type per builtin signature
#define AS1P(p) ((__attribute__((address_space(1))) v4i*)(unsigned long long)(p))
// generic LDS address low 32 bits == LDS byte offset (hardware truncates per aperture spec)
#define AS3P(p) ((__attribute__((address_space(3))) v4i*)(unsigned int)(unsigned long long)(p))
#if __has_builtin(__builtin_amdgcn_s_wait_asynccnt)
#define WAIT_ASYNCCNT(n) __builtin_amdgcn_s_wait_asynccnt(n)
#else
#define WAIT_ASYNCCNT(n) asm volatile("s_wait_asynccnt %0" :: "n"(n) : "memory")
#endif
#endif

// ---------------------------------------------------------------------------
// f32 -> bf16 convert with row/col zero padding (weights + x)
// ---------------------------------------------------------------------------
__global__ __launch_bounds__(256)
void cvt_pad_kernel(const float* __restrict__ src, bf16* __restrict__ dst,
                    int rows, int cols, int rpad, int cpad)
{
    size_t idx = (size_t)blockIdx.x * 256 + threadIdx.x;
    if (idx >= (size_t)rpad * cpad) return;
    int r = (int)(idx / cpad), c = (int)(idx % cpad);
    float v = (r < rows && c < cols) ? src[(size_t)r * cols + c] : 0.f;
    dst[idx] = (bf16)v;
}

__global__ void pad_bias_kernel(const float* __restrict__ src, float* __restrict__ dst,
                                int n, int npad)
{
    int i = blockIdx.x * blockDim.x + threadIdx.x;
    if (i < npad) dst[i] = (i < n) ? src[i] : 0.f;
}

// ---------------------------------------------------------------------------
// encoded = eps * exp(0.5*mu) + mu  (logvar == mu in the reference)
// also zero-initializes GRU hidden state (f32 + bf16 shadow)
// ---------------------------------------------------------------------------
__global__ __launch_bounds__(256)
void reparam_kernel(const float* __restrict__ mu, const float* __restrict__ eps,
                    bf16* __restrict__ x_b, float* __restrict__ h_f,
                    bf16* __restrict__ h_b, int total)
{
    int i = blockIdx.x * 256 + threadIdx.x;
    if (i >= total) return;
    float m = mu[i];
    float enc = eps[i] * __expf(0.5f * m) + m;
    x_b[i] = (bf16)enc;
    h_f[i] = 0.f;
    h_b[i] = (bf16)0.f;
}

// ---------------------------------------------------------------------------
// GRU gate fusion: GI/GH already contain x@Wih^T+bih and h@Whh^T+bhh
// ---------------------------------------------------------------------------
__global__ __launch_bounds__(256)
void gru_gates_kernel(const float* __restrict__ GI, const float* __restrict__ GH,
                      float* __restrict__ h_f, bf16* __restrict__ h_b, int B, int H)
{
    int i = blockIdx.x * 256 + threadIdx.x;
    if (i >= B * H) return;
    int b = i / H, j = i % H;
    size_t base = (size_t)b * 3 * H + j;
    float r = 1.f / (1.f + __expf(-(GI[base]         + GH[base])));
    float z = 1.f / (1.f + __expf(-(GI[base + H]     + GH[base + H])));
    float n = tanhf(GI[base + 2 * H] + r * GH[base + 2 * H]);
    float hn = (1.f - z) * n + z * h_f[i];
    h_f[i] = hn;
    h_b[i] = (bf16)hn;
}

// ---------------------------------------------------------------------------
// WMMA GEMM:  C[M,N] = act(A[M,K] @ W[N,K]^T + bias[N])
// A, W bf16 row-major (K contiguous). 256 threads = 8 waves (4x2),
// each wave owns a 32x32 tile -> 2x2 v_wmma_f32_16x16x32_bf16 accumulators.
// LDS double-buffered; tiles staged via GLOBAL_LOAD_ASYNC_TO_LDS_B128 when
// available (ASYNCcnt pipeline), else reg-prefetch pipeline.
// M%128==0, N%64==0, K%64==0.
// ---------------------------------------------------------------------------
template<int ACT>
__global__ __launch_bounds__(256)
void gemm_bf16_kernel(const bf16* __restrict__ A, const bf16* __restrict__ W,
                      const float* __restrict__ bias,
                      bf16* __restrict__ Cb, int ldcb,
                      float* __restrict__ Cf, int ldcf, int ncf,
                      int M, int N, int K)
{
    __shared__ bf16 sA[2][BM][LDSA];
    __shared__ bf16 sB[2][BN][LDSA];

    const int tid  = threadIdx.x;
    const int wave = tid >> 5;
    const int lane = tid & 31;
    const int wm   = wave & 3;      // wave row  (0..3) -> 32 rows each
    const int wn   = wave >> 2;     // wave col  (0..1) -> 32 cols each
    const int m16  = lane & 15;
    const int hi   = lane >> 4;

    const int m0 = blockIdx.y * BM;
    const int n0 = blockIdx.x * BN;

    v8f acc[2][2] = {};

    auto compute = [&](int buf) {
        #pragma unroll
        for (int kk = 0; kk < BK; kk += 32) {
            union { v16bf v; bf16x8 h[2]; } fa[2], fb[2];
            // A 16x32 bf16 layout: lane holds row m=lane%16,
            //   elems 0..7  = K[hi*8 .. hi*8+7], elems 8..15 = K[16+hi*8 ..]
            #pragma unroll
            for (int i = 0; i < 2; ++i) {
                const bf16* ap = &sA[buf][wm * 32 + i * 16 + m16][kk];
                fa[i].h[0] = *(const bf16x8*)(ap + hi * 8);
                fa[i].h[1] = *(const bf16x8*)(ap + 16 + hi * 8);
            }
            // B 32x16 layout: lane holds col n=lane%16,
            //   elems 0..7  = K[hi*8 .. hi*8+7], elems 8..15 = K[16+hi*8 ..]
            #pragma unroll
            for (int j = 0; j < 2; ++j) {
                const bf16* bp = &sB[buf][wn * 32 + j * 16 + m16][kk];
                fb[j].h[0] = *(const bf16x8*)(bp + hi * 8);
                fb[j].h[1] = *(const bf16x8*)(bp + 16 + hi * 8);
            }
            #pragma unroll
            for (int i = 0; i < 2; ++i)
                #pragma unroll
                for (int j = 0; j < 2; ++j)
                    acc[i][j] = __builtin_amdgcn_wmma_f32_16x16x32_bf16(
                        false, fa[i].v, false, fb[j].v,
                        (short)0, acc[i][j], false, false);
        }
    };

    const int nk = K / BK;

#if USE_ASYNC_COPY
    // Each thread issues 6 async b128 copies per tile (4 for A, 2 for W).
    auto async_tile = [&](int kb, int buf) {
        size_t kbase = (size_t)kb * BK;
        #pragma unroll
        for (int i = 0; i < 4; ++i) {                 // 128x64 A tile: 1024 16B chunks
            int c = tid + i * 256;
            int row = c >> 3, col = (c & 7) * 8;
            __builtin_amdgcn_global_load_async_to_lds_b128(
                AS1P(A + (size_t)(m0 + row) * K + kbase + col),
                AS3P(&sA[buf][row][col]), 0, 0);
        }
        #pragma unroll
        for (int i = 0; i < 2; ++i) {                 // 64x64 W tile: 512 16B chunks
            int c = tid + i * 256;
            int row = c >> 3, col = (c & 7) * 8;
            __builtin_amdgcn_global_load_async_to_lds_b128(
                AS1P(W + (size_t)(n0 + row) * K + kbase + col),
                AS3P(&sB[buf][row][col]), 0, 0);
        }
    };

    async_tile(0, 0);
    for (int kb = 0; kb < nk; ++kb) {
        const int buf = kb & 1;
        if (kb + 1 < nk) {
            async_tile(kb + 1, buf ^ 1);   // in flight while we compute tile kb
            WAIT_ASYNCCNT(6);              // oldest 6 (tile kb) have landed in LDS
        } else {
            WAIT_ASYNCCNT(0);
        }
        __syncthreads();                   // tile kb visible to all waves
        compute(buf);
        __syncthreads();                   // all done reading buf before it is rewritten
    }
#else
    bf16x8 rA[4], rB[2];
    auto fetch = [&](int kb) {
        size_t kbase = (size_t)kb * BK;
        #pragma unroll
        for (int i = 0; i < 4; ++i) {
            int c = tid + i * 256;
            int row = c >> 3, col = (c & 7) * 8;
            rA[i] = *(const bf16x8*)(A + (size_t)(m0 + row) * K + kbase + col);
        }
        #pragma unroll
        for (int i = 0; i < 2; ++i) {
            int c = tid + i * 256;
            int row = c >> 3, col = (c & 7) * 8;
            rB[i] = *(const bf16x8*)(W + (size_t)(n0 + row) * K + kbase + col);
        }
    };
    auto stash = [&](int buf) {
        #pragma unroll
        for (int i = 0; i < 4; ++i) {
            int c = tid + i * 256;
            int row = c >> 3, col = (c & 7) * 8;
            *(bf16x8*)&sA[buf][row][col] = rA[i];
        }
        #pragma unroll
        for (int i = 0; i < 2; ++i) {
            int c = tid + i * 256;
            int row = c >> 3, col = (c & 7) * 8;
            *(bf16x8*)&sB[buf][row][col] = rB[i];
        }
    };

    fetch(0);
    stash(0);
    __syncthreads();
    for (int kb = 0; kb < nk; ++kb) {
        const int buf = kb & 1;
        if (kb + 1 < nk) fetch(kb + 1);
        compute(buf);
        if (kb + 1 < nk) stash(buf ^ 1);
        __syncthreads();
    }
#endif

    // Epilogue: C vgpr r <-> m = r + 8*hi, n = lane%16
    #pragma unroll
    for (int j = 0; j < 2; ++j) {
        const int n = n0 + wn * 32 + j * 16 + m16;
        const float bj = bias ? bias[n] : 0.f;
        #pragma unroll
        for (int i = 0; i < 2; ++i) {
            const int mb = m0 + wm * 32 + i * 16 + hi * 8;
            #pragma unroll
            for (int r = 0; r < 8; ++r) {
                float v = acc[i][j][r] + bj;
                if (ACT == 1) v = v > 0.f ? v : 0.f;
                const int m = mb + r;
                if (Cb) Cb[(size_t)m * ldcb + n] = (bf16)v;
                if (Cf && n < ncf) Cf[(size_t)m * ldcf + n] = v;
            }
        }
    }
}

// ---------------------------------------------------------------------------
static inline void gemm(hipStream_t s, int act,
                        const bf16* A, const bf16* W, const float* bias,
                        bf16* Cb, int ldcb, float* Cf, int ldcf, int ncf,
                        int M, int N, int K)
{
    dim3 g(N / BN, M / BM);
    if (act) gemm_bf16_kernel<1><<<g, 256, 0, s>>>(A, W, bias, Cb, ldcb, Cf, ldcf, ncf, M, N, K);
    else     gemm_bf16_kernel<0><<<g, 256, 0, s>>>(A, W, bias, Cb, ldcb, Cf, ldcf, ncf, M, N, K);
}

extern "C" void kernel_launch(void* const* d_in, const int* in_sizes, int n_in,
                              void* d_out, int out_size, void* d_ws, size_t ws_size,
                              hipStream_t stream)
{
    (void)in_sizes; (void)n_in; (void)out_size; (void)ws_size;
    constexpr int B = 2048, FP = 4860, FPP = 4864, ENC = 512, H = 512;
    constexpr int OUT = 42, OUTP = 64, STEPS = 128;

    const float* x    = (const float*)d_in[0];
    const float* eps  = (const float*)d_in[2];
    const float* We1  = (const float*)d_in[3];  const float* be1  = (const float*)d_in[4];
    const float* We2  = (const float*)d_in[5];  const float* be2  = (const float*)d_in[6];
    const float* We3  = (const float*)d_in[7];  const float* be3  = (const float*)d_in[8];
    const float* We41 = (const float*)d_in[9];  const float* be41 = (const float*)d_in[10];
    const float* Wih  = (const float*)d_in[11]; const float* bih  = (const float*)d_in[12];
    const float* Whh  = (const float*)d_in[13]; const float* bhh  = (const float*)d_in[14];
    const float* Wf1  = (const float*)d_in[15]; const float* bf1  = (const float*)d_in[16];
    const float* Wf2  = (const float*)d_in[17]; const float* bf2  = (const float*)d_in[18];
    float* out = (float*)d_out;

    char* p = (char*)d_ws;
    auto take = [&](size_t bytes) -> char* {
        char* q = p; p += (bytes + 255) & ~(size_t)255; return q;
    };

    bf16*  xpad  = (bf16*)take((size_t)B * FPP * 2);
    bf16*  We1b  = (bf16*)take((size_t)2048 * FPP * 2);
    bf16*  We2b  = (bf16*)take((size_t)1024 * 2048 * 2);
    bf16*  We3b  = (bf16*)take((size_t)512 * 1024 * 2);
    bf16*  We41b = (bf16*)take((size_t)ENC * 512 * 2);
    bf16*  Wihb  = (bf16*)take((size_t)3 * H * ENC * 2);
    bf16*  Whhb  = (bf16*)take((size_t)3 * H * H * 2);
    bf16*  Wf1b  = (bf16*)take((size_t)OUTP * H * 2);
    bf16*  Wf2b  = (bf16*)take((size_t)H * OUTP * 2);
    float* bf1p  = (float*)take((size_t)OUTP * 4);
    bf16*  h1b   = (bf16*)take((size_t)B * 2048 * 2);
    bf16*  h2b   = (bf16*)take((size_t)B * 1024 * 2);
    bf16*  h3b   = (bf16*)take((size_t)B * 512 * 2);
    float* muf   = (float*)take((size_t)B * ENC * 4);
    bf16*  xb    = (bf16*)take((size_t)B * ENC * 2);
    float* hf    = (float*)take((size_t)B * H * 4);
    bf16*  hb    = (bf16*)take((size_t)B * H * 2);
    float* GI    = (float*)take((size_t)B * 3 * H * 4);
    float* GH    = (float*)take((size_t)B * 3 * H * 4);
    bf16*  outb  = (bf16*)take((size_t)B * OUTP * 2);

    auto cvt = [&](const float* src, bf16* dst, int rows, int cols, int rpad, int cpad) {
        size_t total = (size_t)rpad * cpad;
        cvt_pad_kernel<<<(unsigned)((total + 255) / 256), 256, 0, stream>>>(
            src, dst, rows, cols, rpad, cpad);
    };

    // --- precision conversion + padding -------------------------------------
    cvt(x,    xpad,  B,    FP,   B,    FPP);
    cvt(We1,  We1b,  2048, FP,   2048, FPP);
    cvt(We2,  We2b,  1024, 2048, 1024, 2048);
    cvt(We3,  We3b,  512,  1024, 512,  1024);
    cvt(We41, We41b, ENC,  512,  ENC,  512);
    cvt(Wih,  Wihb,  3*H,  ENC,  3*H,  ENC);
    cvt(Whh,  Whhb,  3*H,  H,    3*H,  H);
    cvt(Wf1,  Wf1b,  OUT,  H,    OUTP, H);     // zero rows 42..63 -> padded out cols are 0
    cvt(Wf2,  Wf2b,  H,    OUT,  H,    OUTP);  // zero cols 42..63 match padded K
    pad_bias_kernel<<<1, OUTP, 0, stream>>>(bf1, bf1p, OUT, OUTP);

    // --- VAE encoder --------------------------------------------------------
    gemm(stream, 1, xpad, We1b,  be1,  h1b, 2048, nullptr, 0, 0, B, 2048, FPP);
    gemm(stream, 1, h1b,  We2b,  be2,  h2b, 1024, nullptr, 0, 0, B, 1024, 2048);
    gemm(stream, 1, h2b,  We3b,  be3,  h3b, 512,  nullptr, 0, 0, B, 512, 1024);
    gemm(stream, 0, h3b,  We41b, be41, nullptr, 0, muf, ENC, ENC, B, ENC, 512);

    reparam_kernel<<<(B * ENC) / 256, 256, 0, stream>>>(muf, eps, xb, hf, hb, B * ENC);

    // --- GRU decoder (sequential over 128 steps) ----------------------------
    for (int t = 0; t < STEPS; ++t) {
        gemm(stream, 0, xb, Wihb, bih, nullptr, 0, GI, 3*H, 3*H, B, 3*H, ENC);
        gemm(stream, 0, hb, Whhb, bhh, nullptr, 0, GH, 3*H, 3*H, B, 3*H, H);
        gru_gates_kernel<<<(B * H) / 256, 256, 0, stream>>>(GI, GH, hf, hb, B, H);
        // out = relu(h_new @ Wf1^T + bf1): f32 slice into d_out[b, t, :42],
        // bf16 zero-padded copy (cols 42..63 exactly 0) for the next GEMM
        gemm(stream, 1, hb, Wf1b, bf1p, outb, OUTP,
             out + (size_t)t * OUT, STEPS * OUT, OUT, B, OUTP, H);
        // x_new = relu(out @ Wf2^T + bf2), K padded to 64
        gemm(stream, 1, outb, Wf2b, bf2, xb, ENC, nullptr, 0, 0, B, ENC, OUTP);
    }
}
